// CrossModalMoELayer_35880156791506
// MI455X (gfx1250) — matrix operand order repaired and checked by reading, not compiled, not run on hardware
//
#include <hip/hip_runtime.h>
#include <hip/hip_bf16.h>

// ---------------------------------------------------------------------------
// CDNA5 (gfx1250) implementation of CrossModalMoELayer.
// Dominant cost: dense soft-MoE GEMMs (~446 GFLOP) -> bf16 WMMA
// (v_wmma_f32_16x16x32_bf16), fp32 accumulate, double-buffered LDS pipeline.
// Attention K/V staging uses GLOBAL_LOAD_ASYNC_TO_LDS (ASYNCcnt path).
// ---------------------------------------------------------------------------

typedef __attribute__((ext_vector_type(16))) __bf16        v16bf;
typedef __attribute__((ext_vector_type(2)))  __bf16        v2bf;
typedef __attribute__((ext_vector_type(8)))  float         v8f;
typedef __attribute__((ext_vector_type(4)))  unsigned int  u32x4;
typedef __attribute__((ext_vector_type(2)))  unsigned int  u32x2;
typedef __attribute__((ext_vector_type(4)))  int           i32x4;
typedef __attribute__((address_space(3)))    i32x4         lds_i32x4;

#define DEV static __device__ __forceinline__

#if defined(__has_builtin)
#  if __has_builtin(__builtin_amdgcn_cvt_pk_bf16_f32)
#    define HAVE_CVT_PK_BF16 1
#  endif
#  if __has_builtin(__builtin_amdgcn_global_load_async_to_lds_b128)
#    define HAVE_ASYNC_COPY 1
#  endif
#  if __has_builtin(__builtin_amdgcn_s_wait_asynccnt)
#    define HAVE_WAIT_ASYNC 1
#  endif
#endif

// Problem constants (match reference)
constexpr int Bc = 4, Qc = 256, Ic = 576, Tc = 512;
constexpr int Hc = 1024, NHc = 16, DHc = 64, DIc = 4096, Ec = 8;

// GEMM tiling: 256 threads = 8 waves (wave32). Macro tile 256x128, K-step 32.
// Waves arranged 4x2; each wave owns a 64x64 tile = 4x4 WMMA 16x16 frags
// -> 16 v_wmma per wave per K-step against 16 ds_load_b128 (ratio 1.0).
constexpr int BM = 256, BN = 128, BK = 32, NTHREADS = 256;

DEV unsigned bf16_bits(float f) {
  unsigned u = __builtin_bit_cast(unsigned, f);
  return (u + 0x7FFFu + ((u >> 16) & 1u)) >> 16;  // RNE
}
DEV unsigned pack2_bf16(float a, float b) {
#if defined(HAVE_CVT_PK_BF16)
  v2bf r = __builtin_amdgcn_cvt_pk_bf16_f32(a, b);
  return __builtin_bit_cast(unsigned, r);
#else
  return bf16_bits(a) | (bf16_bits(b) << 16);
#endif
}
DEV float gelu_tanh(float x) {
  return 0.5f * x * (1.0f + tanhf(0.7978845608f * (x + 0.044715f * x * x * x)));
}

union Frag16 {           // one WMMA A/B operand: 16 bf16 per lane = 8 VGPRs
  v16bf v;
  u32x4 q[2];
};

// ---------------------------------------------------------------------------
// WMMA GEMM: C[M,N] = A[M,K] @ W[N,K]^T  (+bias along N), fused epilogues.
// MODE 0: C = Y + bias
// MODE 1: C = extra + Y + bias                  (residual add)
// MODE 2: C = gelu(Y + bias)                    (expert up-proj)
// MODE 3: C += extra[m*E + expert] * (Y + bias) (prob-weighted expert mix)
// Requires M % 256 == 0, N % 128 == 0, K % 32 == 0 (true for all call sites).
// ---------------------------------------------------------------------------
template <int MODE>
__global__ __launch_bounds__(NTHREADS)
void gemm_bf16_wmma(const float* __restrict__ A, const float* __restrict__ W,
                    const float* __restrict__ bias, const float* __restrict__ extra,
                    float* __restrict__ C, int M, int N, int K, int expert) {
  const int nTile = blockIdx.x * BN;
  const int mTile = blockIdx.y * BM;
  const int tid  = threadIdx.x;
  const int lane = tid & 31;
  const int wave = tid >> 5;     // 0..7
  const int wm   = wave & 3;     // wave row 0..3 (64 rows each)
  const int wn   = wave >> 2;    // wave col 0..1 (64 cols each)

  // double-buffered bf16 staging: (32 + 16) KB
  __shared__ alignas(16) __bf16 As[2][BM][BK];
  __shared__ alignas(16) __bf16 Bs[2][BN][BK];

  const v8f vzero = {0.f, 0.f, 0.f, 0.f, 0.f, 0.f, 0.f, 0.f};
  v8f acc[4][4];
#pragma unroll
  for (int i = 0; i < 4; ++i)
#pragma unroll
    for (int j = 0; j < 4; ++j) acc[i][j] = vzero;

  const float* Abase = A + (size_t)mTile * K;
  const float* Wbase = W + (size_t)nTile * K;

  const int lr = tid >> 3;        // 0..31 : row within 32-row load slab
  const int lc = (tid & 7) * 4;   // 0,4,...,28 : starting K column (float4)

  // staging registers: A 8 slabs (256 rows), B 4 slabs (128 rows)
  float4 rA[8], rB[4];

  auto load_tile = [&](int k0) {
#pragma unroll
    for (int i = 0; i < 8; ++i)
      rA[i] = *(const float4*)(Abase + (size_t)(lr + 32 * i) * K + k0 + lc);
#pragma unroll
    for (int i = 0; i < 4; ++i)
      rB[i] = *(const float4*)(Wbase + (size_t)(lr + 32 * i) * K + k0 + lc);
  };
  auto store_tile = [&](int buf) {
#pragma unroll
    for (int i = 0; i < 8; ++i) {
      u32x2 p; p.x = pack2_bf16(rA[i].x, rA[i].y); p.y = pack2_bf16(rA[i].z, rA[i].w);
      *(u32x2*)&As[buf][lr + 32 * i][lc] = p;
    }
#pragma unroll
    for (int i = 0; i < 4; ++i) {
      u32x2 p; p.x = pack2_bf16(rB[i].x, rB[i].y); p.y = pack2_bf16(rB[i].z, rB[i].w);
      *(u32x2*)&Bs[buf][lr + 32 * i][lc] = p;
    }
  };

  // prologue: stage K-tile 0
  load_tile(0);
  store_tile(0);
  __syncthreads();

  const int row16 = lane & 15;
  const int kh    = lane >> 4;   // lane half selects K grouping

  int cur = 0;
  for (int k0 = 0; k0 < K; k0 += BK) {
    const int  nxt      = cur ^ 1;
    const bool has_next = (k0 + BK) < K;
    if (has_next) load_tile(k0 + BK);            // issue globals early (hidden by WMMAs)
    if (k0 + 2 * BK < K) {                       // CDNA5 global_prefetch_b8, next-next tile
      __builtin_prefetch(Abase + (size_t)lr * K + k0 + 2 * BK + lc, 0, 0);
      __builtin_prefetch(Wbase + (size_t)lr * K + k0 + 2 * BK + lc, 0, 0);
    }

    // ---- fragments from LDS (ISA 7.12.2 layouts) ----
    Frag16 a[4], b[4];
#pragma unroll
    for (int tm = 0; tm < 4; ++tm) {
      // A 16x32 bf16: lanes 0-15 hold K{0..7,16..23}; lanes 16-31 K{8..15,24..31}
      const __bf16* p = &As[cur][wm * 64 + tm * 16 + row16][kh * 8];
      a[tm].q[0] = *(const u32x4*)p;          // K  8kh .. 8kh+7
      a[tm].q[1] = *(const u32x4*)(p + 16);   // K 16+8kh .. 23+8kh
    }
#pragma unroll
    for (int tn = 0; tn < 4; ++tn) {
      // B 32x16 bf16: lane = column; lanes 0-15 K{0..7,16..23}; lanes 16-31 K{8..15,24..31}
      const __bf16* p = &Bs[cur][wn * 64 + tn * 16 + row16][kh * 8];
      b[tn].q[0] = *(const u32x4*)p;
      b[tn].q[1] = *(const u32x4*)(p + 16);
    }
#pragma unroll
    for (int tm = 0; tm < 4; ++tm)
#pragma unroll
      for (int tn = 0; tn < 4; ++tn)
        acc[tm][tn] = __builtin_amdgcn_wmma_f32_16x16x32_bf16(
            false, a[tm].v, false, b[tn].v, (short)0, acc[tm][tn], false, false);

    if (has_next) store_tile(nxt);               // s_wait_loadcnt lands after the math
    __syncthreads();                             // one barrier per K-step (ping-pong)
    cur = nxt;
  }

  // ---- epilogue (C 16x16 f32: VGPR r -> M = r + 8*(lane>>4), N = lane&15) ----
  const int nl    = lane & 15;
  const int mhalf = (lane >> 4) * 8;
#pragma unroll
  for (int tm = 0; tm < 4; ++tm) {
#pragma unroll
    for (int tn = 0; tn < 4; ++tn) {
      const int ng  = nTile + wn * 64 + tn * 16 + nl;
      const int mg0 = mTile + wm * 64 + tm * 16 + mhalf;
      const float bval = bias ? bias[ng] : 0.f;
#pragma unroll
      for (int r = 0; r < 8; ++r) {
        const int mg = mg0 + r;
        const size_t idx = (size_t)mg * N + ng;
        const float y = acc[tm][tn][r] + bval;
        if constexpr (MODE == 0) {
          C[idx] = y;
        } else if constexpr (MODE == 1) {
          C[idx] = extra[idx] + y;
        } else if constexpr (MODE == 2) {
          C[idx] = gelu_tanh(y);
        } else {  // MODE == 3
          C[idx] += extra[(size_t)mg * Ec + expert] * y;
        }
      }
    }
  }
}

// ---------------------------------------------------------------------------
// Online-softmax attention. One thread per query row (blockDim == Lq == 256),
// one block per (batch, head). K/V streamed through LDS in 64-key chunks,
// staged with async global->LDS copies when available (ASYNCcnt path).
// ---------------------------------------------------------------------------
constexpr int KVPAD = DHc + 4;   // row stride 272B: keeps 16B alignment for b128

__global__ __launch_bounds__(256)
void attention_kernel(const float* __restrict__ Qb, const float* __restrict__ Kb,
                      const float* __restrict__ Vb, float* __restrict__ Ob,
                      int Lq, int Lk, int q_stride, int kv_stride, float scale) {
  const int bh = blockIdx.x;
  const int b = bh / NHc, h = bh % NHc;
  const int t = threadIdx.x;  // query row

  __shared__ alignas(16) float Ks[64][KVPAD];
  __shared__ alignas(16) float Vs[64][KVPAD];

  const float* qp = Qb + ((size_t)(b * Lq + t)) * q_stride + h * DHc;
  float qreg[DHc];
#pragma unroll
  for (int d = 0; d < DHc; ++d) qreg[d] = qp[d];

  float m = -3.0e38f, l = 0.f;
  float oacc[DHc];
#pragma unroll
  for (int d = 0; d < DHc; ++d) oacc[d] = 0.f;

  const int lr = t >> 2;         // 0..63 key row
  const int lc = (t & 3) * 16;   // 16 floats (64B) per thread

  for (int c0 = 0; c0 < Lk; c0 += 64) {
    const float* kp = Kb + ((size_t)(b * Lk + c0 + lr)) * kv_stride + h * DHc + lc;
    const float* vp = Vb + ((size_t)(b * Lk + c0 + lr)) * kv_stride + h * DHc + lc;
#if defined(HAVE_ASYNC_COPY) && defined(HAVE_WAIT_ASYNC)
#pragma unroll
    for (int j = 0; j < 4; ++j) {   // 4x b128 per thread for K, 4x for V
      __builtin_amdgcn_global_load_async_to_lds_b128(
          (i32x4*)(uintptr_t)(kp + j * 4),
          (lds_i32x4*)(uintptr_t)&Ks[lr][lc + j * 4], 0, 0);
      __builtin_amdgcn_global_load_async_to_lds_b128(
          (i32x4*)(uintptr_t)(vp + j * 4),
          (lds_i32x4*)(uintptr_t)&Vs[lr][lc + j * 4], 0, 0);
    }
    __builtin_amdgcn_s_wait_asynccnt(0);
#else
#pragma unroll
    for (int j = 0; j < 16; ++j) { Ks[lr][lc + j] = kp[j]; Vs[lr][lc + j] = vp[j]; }
#endif
    __syncthreads();

    const int cnt = (Lk - c0 < 64) ? (Lk - c0) : 64;
    for (int kk = 0; kk < cnt; ++kk) {
      float s = 0.f;
#pragma unroll
      for (int d = 0; d < DHc; ++d) s += qreg[d] * Ks[kk][d];
      s *= scale;
      const float mNew = fmaxf(m, s);
      const float corr = __expf(m - mNew);
      const float p    = __expf(s - mNew);
      l = l * corr + p;
#pragma unroll
      for (int d = 0; d < DHc; ++d) oacc[d] = oacc[d] * corr + p * Vs[kk][d];
      m = mNew;
    }
    __syncthreads();
  }

  const float inv = 1.f / l;
  float* op = Ob + ((size_t)(b * Lq + t)) * Hc + h * DHc;
#pragma unroll
  for (int d = 0; d < DHc; ++d) op[d] = oacc[d] * inv;
}

// ---------------------------------------------------------------------------
// LayerNorm over H=1024 per row. One block per row.
// ---------------------------------------------------------------------------
__global__ __launch_bounds__(256)
void ln_kernel(const float* __restrict__ x, const float* __restrict__ g,
               const float* __restrict__ bta, float* __restrict__ y) {
  const int row = blockIdx.x, tid = threadIdx.x;
  const float* xp = x + (size_t)row * Hc;
  float s = 0.f, s2 = 0.f;
  for (int h = tid; h < Hc; h += 256) { const float v = xp[h]; s += v; s2 += v * v; }
  __shared__ float rs[256], rs2[256];
  rs[tid] = s; rs2[tid] = s2; __syncthreads();
  for (int off = 128; off > 0; off >>= 1) {
    if (tid < off) { rs[tid] += rs[tid + off]; rs2[tid] += rs2[tid + off]; }
    __syncthreads();
  }
  const float mu = rs[0] * (1.f / Hc);
  const float var = rs2[0] * (1.f / Hc) - mu * mu;
  const float rstd = rsqrtf(var + 1e-5f);
  float* yp = y + (size_t)row * Hc;
  for (int h = tid; h < Hc; h += 256) yp[h] = (xp[h] - mu) * rstd * g[h] + bta[h];
}

// Column means: out[b*H + h] = mean_r x[(b*rows + r)*H + h]
__global__ void mean_kernel(const float* __restrict__ x, float* __restrict__ out, int rows) {
  const int idx = blockIdx.x * blockDim.x + threadIdx.x;
  if (idx >= Bc * Hc) return;
  const int b = idx / Hc, h = idx % Hc;
  float s = 0.f;
  for (int r = 0; r < rows; ++r) s += x[((size_t)b * rows + r) * Hc + h];
  out[idx] = s / (float)rows;
}

// Gate logits + softmax over E=8: concat([X, ctx]) @ Wg^T + bg
__global__ void gate_kernel(const float* __restrict__ X, const float* __restrict__ ctx,
                            const float* __restrict__ Wg, const float* __restrict__ bg,
                            float* __restrict__ probs, int tokens, int tok_per_b) {
  const int t = blockIdx.x * blockDim.x + threadIdx.x;
  if (t >= tokens) return;
  const int b = t / tok_per_b;
  float acc[Ec];
#pragma unroll
  for (int e = 0; e < Ec; ++e) acc[e] = bg[e];
  const float* xp = X + (size_t)t * Hc;
  const float* cp = ctx + (size_t)b * Hc;
  for (int h = 0; h < Hc; ++h) {
    const float xv = xp[h], cv = cp[h];
#pragma unroll
    for (int e = 0; e < Ec; ++e)
      acc[e] += xv * Wg[(size_t)e * 2 * Hc + h] + cv * Wg[(size_t)e * 2 * Hc + Hc + h];
  }
  float mx = acc[0];
#pragma unroll
  for (int e = 1; e < Ec; ++e) mx = fmaxf(mx, acc[e]);
  float s = 0.f;
#pragma unroll
  for (int e = 0; e < Ec; ++e) { acc[e] = __expf(acc[e] - mx); s += acc[e]; }
  const float inv = 1.f / s;
#pragma unroll
  for (int e = 0; e < Ec; ++e) probs[(size_t)t * Ec + e] = acc[e] * inv;
}

__global__ void copy_kernel(const float* __restrict__ a, float* __restrict__ o, int n) {
  const int i = blockIdx.x * blockDim.x + threadIdx.x;
  if (i < n) o[i] = a[i];
}

// ---------------------------------------------------------------------------
extern "C" void kernel_launch(void* const* d_in, const int* in_sizes, int n_in,
                              void* d_out, int out_size, void* d_ws, size_t ws_size,
                              hipStream_t stream) {
  (void)in_sizes; (void)n_in; (void)out_size; (void)ws_size;
  const float* query    = (const float*)d_in[0];
  const float* image    = (const float*)d_in[1];
  const float* text     = (const float*)d_in[2];
  const float* sa_in_w  = (const float*)d_in[3];
  const float* sa_in_b  = (const float*)d_in[4];
  const float* sa_out_w = (const float*)d_in[5];
  const float* sa_out_b = (const float*)d_in[6];
  const float* ca_in_w  = (const float*)d_in[7];
  const float* ca_in_b  = (const float*)d_in[8];
  const float* ca_out_w = (const float*)d_in[9];
  const float* ca_out_b = (const float*)d_in[10];
  const float* img_gw   = (const float*)d_in[11];
  const float* img_gb   = (const float*)d_in[12];
  const float* txt_gw   = (const float*)d_in[13];
  const float* txt_gb   = (const float*)d_in[14];
  const float* w1       = (const float*)d_in[15];
  const float* b1       = (const float*)d_in[16];
  const float* w2       = (const float*)d_in[17];
  const float* b2       = (const float*)d_in[18];
  const float* nq_g     = (const float*)d_in[19];
  const float* nq_b     = (const float*)d_in[20];
  const float* nc_g     = (const float*)d_in[21];
  const float* nc_b     = (const float*)d_in[22];
  const float* nf_g     = (const float*)d_in[23];
  const float* nf_b     = (const float*)d_in[24];

  const int BQ = Bc * Qc;   // 1024 text-side tokens  (4 x 256-row tiles)
  const int BI = Bc * Ic;   // 2304 image tokens      (9 x 256-row tiles)

  // workspace carve (fp32)
  float* ws = (float*)d_ws;
  size_t o = 0;
  float* q_norm = ws + o; o += (size_t)BQ * Hc;
  float* qkv    = ws + o; o += (size_t)BQ * 3 * Hc;
  float* attn   = ws + o; o += (size_t)BQ * Hc;
  float* qt1    = ws + o; o += (size_t)BQ * Hc;
  float* cnorm  = ws + o; o += (size_t)BQ * Hc;
  float* q_ca   = ws + o; o += (size_t)BQ * Hc;
  float* kv_img = ws + o; o += (size_t)BI * 2 * Hc;
  float* qt2    = ws + o; o += (size_t)BQ * Hc;
  float* ffin   = ws + o; o += (size_t)BQ * Hc;
  float* ictx   = ws + o; o += (size_t)Bc * Hc;
  float* tctx   = ws + o; o += (size_t)Bc * Hc;
  float* ptxt   = ws + o; o += (size_t)BQ * Ec;
  float* pimg   = ws + o; o += (size_t)BI * Ec;
  float* hbuf   = ws + o; o += (size_t)BI * DIc;   // reused per expert

  float* out_txt = (float*)d_out;
  float* out_img = out_txt + (size_t)BQ * Hc;

  const float scale = 0.125f;  // 1/sqrt(DH=64)

#define GEMM(MODE, A_, W_, BIAS_, EXTRA_, C_, M_, N_, K_, E_)                       \
  gemm_bf16_wmma<MODE><<<dim3((N_) / BN, (M_) / BM), NTHREADS, 0, stream>>>(        \
      A_, W_, BIAS_, EXTRA_, C_, M_, N_, K_, E_)

  // ---- self-attention block ----
  ln_kernel<<<BQ, 256, 0, stream>>>(query, nq_g, nq_b, q_norm);
  GEMM(0, q_norm, sa_in_w, sa_in_b, nullptr, qkv, BQ, 3 * Hc, Hc, 0);
  attention_kernel<<<Bc * NHc, 256, 0, stream>>>(
      qkv, qkv + Hc, qkv + 2 * Hc, attn, Qc, Qc, 3 * Hc, 3 * Hc, scale);
  GEMM(1, attn, sa_out_w, sa_out_b, query, qt1, BQ, Hc, Hc, 0);

  // ---- cross-attention block ----
  ln_kernel<<<BQ, 256, 0, stream>>>(qt1, nc_g, nc_b, cnorm);
  GEMM(0, cnorm, ca_in_w, ca_in_b, nullptr, q_ca, BQ, Hc, Hc, 0);
  GEMM(0, image, ca_in_w + (size_t)Hc * Hc, ca_in_b + Hc, nullptr, kv_img, BI, 2 * Hc, Hc, 0);
  attention_kernel<<<Bc * NHc, 256, 0, stream>>>(
      q_ca, kv_img, kv_img + Hc, attn, Qc, Ic, Hc, 2 * Hc, scale);
  GEMM(1, attn, ca_out_w, ca_out_b, qt1, qt2, BQ, Hc, Hc, 0);

  // ---- gating ----
  mean_kernel<<<(Bc * Hc + 255) / 256, 256, 0, stream>>>(image, ictx, Ic);
  mean_kernel<<<(Bc * Hc + 255) / 256, 256, 0, stream>>>(text, tctx, Tc);
  gate_kernel<<<(BQ + 255) / 256, 256, 0, stream>>>(qt2, ictx, txt_gw, txt_gb, ptxt, BQ, Qc);
  gate_kernel<<<(BI + 255) / 256, 256, 0, stream>>>(image, tctx, img_gw, img_gb, pimg, BI, Ic);

  // ---- soft MoE ----
  ln_kernel<<<BQ, 256, 0, stream>>>(qt2, nf_g, nf_b, ffin);
  copy_kernel<<<(BQ * Hc + 255) / 256, 256, 0, stream>>>(qt2, out_txt, BQ * Hc);
  copy_kernel<<<(BI * Hc + 255) / 256, 256, 0, stream>>>(image, out_img, BI * Hc);

  for (int e = 0; e < Ec; ++e) {
    const float* w1e = w1 + (size_t)e * DIc * Hc;
    const float* b1e = b1 + (size_t)e * DIc;
    const float* w2e = w2 + (size_t)e * Hc * DIc;
    const float* b2e = b2 + (size_t)e * Hc;
    // text tokens
    GEMM(2, ffin, w1e, b1e, nullptr, hbuf, BQ, DIc, Hc, 0);
    GEMM(3, hbuf, w2e, b2e, ptxt, out_txt, BQ, Hc, DIc, e);
    // image tokens (no norm on image path per reference)
    GEMM(2, image, w1e, b1e, nullptr, hbuf, BI, DIc, Hc, 0);
    GEMM(3, hbuf, w2e, b2e, pimg, out_img, BI, Hc, DIc, e);
  }
#undef GEMM
}
